// Model_36867999269139
// MI455X (gfx1250) — matrix operand order repaired and checked by reading, not compiled, not run on hardware
//
#include <hip/hip_runtime.h>
#include <hip/hip_bf16.h>

typedef _Float16 f16;
typedef __attribute__((ext_vector_type(16))) _Float16 v16h;
typedef __attribute__((ext_vector_type(8)))  float    v8f;
typedef unsigned int v4u __attribute__((ext_vector_type(4)));
typedef int          v8i __attribute__((ext_vector_type(8)));
typedef int          v4i __attribute__((ext_vector_type(4)));

#define NIMG 1024

// ---------------------------------------------------------------------------
// Prep: fold BN into scale/bias; weights f32 OIHW -> f16 [Cout][Kpad] with
// K ordered (ky, kx, ci) so that ci is fastest (matches NHWC activations).
// ---------------------------------------------------------------------------
__global__ void prep_layer_kernel(const float* __restrict__ w,
                                  const float* __restrict__ b,
                                  const float* __restrict__ beta,
                                  const float* __restrict__ gamma,
                                  const float* __restrict__ mean,
                                  const float* __restrict__ var,
                                  f16* __restrict__ wout,
                                  float* __restrict__ scale,
                                  float* __restrict__ bias,
                                  int Cout, int Cin, int KH, int KW, int Kpad)
{
    int tid    = blockIdx.x * blockDim.x + threadIdx.x;
    int stride = gridDim.x * blockDim.x;
    int Kreal  = Cin * KH * KW;
    if (tid < Cout) {
        float s    = gamma[tid] * rsqrtf(var[tid] + 1e-5f);
        scale[tid] = s;
        bias[tid]  = (b[tid] - mean[tid]) * s + beta[tid];
    }
    int total = Cout * Kpad;
    for (int t = tid; t < total; t += stride) {
        int co = t / Kpad;
        int kp = t - co * Kpad;
        f16 v = (f16)0.0f;
        if (kp < Kreal) {
            int ci   = kp % Cin;
            int rest = kp / Cin;
            int kx   = rest % KW;
            int ky   = rest / KW;
            v = (f16)w[(((size_t)co * Cin + ci) * KH + ky) * KW + kx];
        }
        wout[t] = v;
    }
}

// ---------------------------------------------------------------------------
// Windowed input -> padded NHWC f16 (NIMG, 86, 17, 1); halo pre-zeroed.
// ---------------------------------------------------------------------------
__global__ void window_kernel(const float* __restrict__ x, f16* __restrict__ out)
{
    int idx = blockIdx.x * blockDim.x + threadIdx.x;
    const int total = NIMG * 84 * 15;
    if (idx >= total) return;
    int wi = idx % 15;
    int t2 = idx / 15;
    int f  = t2 % 84;
    int bt = t2 / 84;
    int b  = bt >> 7;
    int t  = bt & 127;
    int tt = t + wi - 7;
    float v = (tt >= 0 && tt < 128) ? x[((size_t)(b * 128 + tt)) * 84 + f] : 0.0f;
    out[(size_t)(bt * 86 + (f + 1)) * 17 + (wi + 1)] = (f16)v;
}

// ---------------------------------------------------------------------------
// TDM: DMA a 2D weight tile (32 rows x KBLK f16, row stride Kpad elements)
// from global memory into LDS. Descriptor per CDNA5 ISA D# (groups 0/1;
// remaining groups zero for a 2D tensor).
// ---------------------------------------------------------------------------
__device__ __forceinline__
void tdm_load_weight_tile(const f16* gsrc, unsigned int lds_byte_addr,
                          int kblk, int kpad)
{
    unsigned long long ga = (unsigned long long)(uintptr_t)gsrc;
    v4u g0;
    g0[0] = 1u;                                        // count=1, user mode
    g0[1] = lds_byte_addr;                             // lds_addr
    g0[2] = (unsigned int)(ga & 0xffffffffull);        // global_addr[31:0]
    g0[3] = (unsigned int)((ga >> 32) & 0x01ffffffull) // global_addr[56:32]
          | (2u << 30);                                // type=2 ("image")

    v8i g1;
    g1[0] = (int)(1u << 16);                           // data_size=1 (2 bytes)
    g1[1] = (int)(((unsigned)kpad & 0xffffu) << 16);   // tensor_dim0[15:0]
    g1[2] = (int)((((unsigned)kpad >> 16) & 0xffffu)   // tensor_dim0[31:16]
          | (32u << 16));                              // tensor_dim1[15:0]=32
    g1[3] = (int)(((unsigned)kblk & 0xffffu) << 16);   // tile_dim0=KBLK
    g1[4] = 32;                                        // tile_dim1=32 rows
    g1[5] = kpad;                                      // tensor_dim0_stride lo
    g1[6] = 0;                                         // stride hi / dim1_stride lo
    g1[7] = 0;

    v4i gz4 = {0, 0, 0, 0};
    v8i gz8 = {0, 0, 0, 0, 0, 0, 0, 0};
    __builtin_amdgcn_tensor_load_to_lds(g0, g1, gz4, gz4, gz8, 0);
}

// ---------------------------------------------------------------------------
// Implicit-GEMM conv + fused BN + ReLU.
//  - All convs are "valid" convs on halo-padded buffers: unconditional gathers.
//  - Weight K-blocks DMA'd into LDS by the Tensor Data Mover (wave 0 issues,
//    s_wait_tensorcnt, block barrier publishes); A frags read as ds_load_b128.
//  - Per wave: 32(Cout) x 32(pos); per 32-K chunk: 4x v_wmma_f32_16x16x32_f16.
//  - gather addr = pbase[j] (wave-uniform) + koff (lane-uniform per chunk).
// ---------------------------------------------------------------------------
template<int Cin, int IH, int IW, int Cout, int OH, int OW,
         int OHB, int OWB, int OOFS, int KH, int KW, int Kpad>
__global__ __launch_bounds__(256)
void conv_wmma_kernel(const f16* __restrict__ in, f16* __restrict__ out,
                      const f16* __restrict__ wbuf,
                      const float* __restrict__ scale,
                      const float* __restrict__ bias)
{
    constexpr int OHOW  = OH * OW;
    constexpr int Kreal = Cin * KH * KW;
    constexpr int KBLK  = (Kpad <= 576) ? Kpad : 576;   // divides all Kpads
    static_assert(Kpad % KBLK == 0, "KBLK must divide Kpad");
    constexpr int NCHUNK = KBLK / 32;

    __shared__ f16 smem[32 * KBLK];

    const int tid  = threadIdx.x;
    const int lane = tid & 31;
    const int wave = tid >> 5;
    const int n0   = (blockIdx.x * 8 + wave) * 32;  // 32 output positions
    const int m0   = blockIdx.y * 32;               // 32 output channels

    // Incremental decode of 32 consecutive positions (wave-uniform -> SALU)
    int pbase[32];
    {
        int n   = n0;
        int img = n / OHOW;
        int r   = n - img * OHOW;
        int y   = r / OW;
        int x   = r - y * OW;
        int pb  = ((img * IH + y) * IW + x) * Cin;
#pragma unroll
        for (int j = 0; j < 32; ++j) {
            pbase[j] = pb;
            ++x; pb += Cin;
            if (x == OW) {
                x = 0; ++y; pb += (IW - OW) * Cin;
                if (y == OH) { y = 0; pb += (IH - OH) * IW * Cin; }
            }
        }
    }

    const int lm = lane & 15;
    const int lh = lane >> 4;
    const unsigned int smem_base = (unsigned int)(uintptr_t)(&smem[0]);

    v8f acc00 = {}, acc01 = {}, acc10 = {}, acc11 = {};

    union FragA { v16h v; uint4 q[2]; };
    union FragB { v16h v; f16  e[16]; };

    for (int kb = 0; kb < Kpad; kb += KBLK) {
        __syncthreads();   // protect previous iteration's LDS reads
        if (wave == 0) {
            tdm_load_weight_tile(wbuf + (size_t)m0 * Kpad + kb,
                                 smem_base, KBLK, Kpad);
            __builtin_amdgcn_s_wait_tensorcnt(0);
        }
        __syncthreads();   // publish DMA'd weights to all waves

        const f16* srow0 = &smem[lm * KBLK];
        const f16* srow1 = &smem[(lm + 16) * KBLK];

        for (int c = 0; c < NCHUNK; ++c) {
            const int kc = c * 32;

            // ---- A fragments from LDS (ISA K-split layout as 2x16B runs) ----
            FragA A0, A1;
            A0.q[0] = *(const uint4*)(srow0 + kc + 8 * lh);
            A0.q[1] = *(const uint4*)(srow0 + kc + 16 + 8 * lh);
            A1.q[0] = *(const uint4*)(srow1 + kc + 8 * lh);
            A1.q[1] = *(const uint4*)(srow1 + kc + 16 + 8 * lh);

            // ---- per-element K offsets for this chunk (table order, matches A) ----
            int koffs[16];
#pragma unroll
            for (int j = 0; j < 16; ++j) {
                const int k    = kb + kc + ((j < 8) ? (8 * lh + j) : (16 + 8 * lh + (j - 8)));
                const int ci   = k % Cin;
                const int rest = k / Cin;
                const int kx   = rest % KW;
                const int ky   = rest / KW;
                int koff = (ky * IW + kx) * Cin + ci;
                if constexpr (Kpad != Kreal) {
                    // padded K rows: A weights are zero; clamp to a safe address
                    if (k >= Kreal) koff = 0;
                }
                koffs[j] = koff;
            }

            // ---- B fragment 0 (positions 0..15) + 2 WMMA ----
            FragB B;
#pragma unroll
            for (int j = 0; j < 16; ++j)
                B.e[j] = in[(size_t)(pbase[lm] + koffs[j])];
            acc00 = __builtin_amdgcn_wmma_f32_16x16x32_f16(false, A0.v, false, B.v, (short)0, acc00, false, false);
            acc10 = __builtin_amdgcn_wmma_f32_16x16x32_f16(false, A1.v, false, B.v, (short)0, acc10, false, false);

            // ---- B fragment 1 (positions 16..31) + 2 WMMA ----
#pragma unroll
            for (int j = 0; j < 16; ++j)
                B.e[j] = in[(size_t)(pbase[lm + 16] + koffs[j])];
            acc01 = __builtin_amdgcn_wmma_f32_16x16x32_f16(false, A0.v, false, B.v, (short)0, acc01, false, false);
            acc11 = __builtin_amdgcn_wmma_f32_16x16x32_f16(false, A1.v, false, B.v, (short)0, acc11, false, false);
        }
    }

    // ---- Epilogue: scale/bias + ReLU, store NHWC f16 (interior of out buf) ----
    v8f* accs[4] = { &acc00, &acc01, &acc10, &acc11 };
#pragma unroll
    for (int s = 0; s < 2; ++s) {
#pragma unroll
        for (int t = 0; t < 2; ++t) {
            v8f& a   = *accs[s * 2 + t];
            int  n   = n0 + t * 16 + lm;
            int  img = n / OHOW;
            int  r   = n - img * OHOW;
            int  y   = r / OW;
            int  x   = r - y * OW;
            size_t base = ((size_t)(img * OHB + y + OOFS) * OWB + x + OOFS) * Cout;
#pragma unroll
            for (int rr = 0; rr < 8; ++rr) {
                int   m = m0 + s * 16 + rr + 8 * lh;  // D layout: M = rr + 8*half
                float v = a[rr] * scale[m] + bias[m];
                v = fmaxf(v, 0.0f);
                out[base + m] = (f16)v;
            }
        }
    }
}

// ---------------------------------------------------------------------------
// 2x1 max-pool over H (NHWC); input may be halo-padded (IOFS interior offset)
// ---------------------------------------------------------------------------
template<int C, int Hlog, int Wlog, int IHB, int IWB, int IOFS>
__global__ void pool_kernel(const f16* __restrict__ in, f16* __restrict__ out)
{
    constexpr int Ho = Hlog / 2;
    const int total  = NIMG * Ho * Wlog * C;
    int idx = blockIdx.x * blockDim.x + threadIdx.x;
    if (idx >= total) return;
    int c = idx % C;
    int r = idx / C;
    int x = r % Wlog;
    r /= Wlog;
    int y = r % Ho;
    int n = r / Ho;
    const f16* p = in + ((size_t)(n * IHB + 2 * y + IOFS) * IWB + x + IOFS) * C + c;
    f16 a = p[0];
    f16 b = p[(size_t)IWB * C];
    out[idx] = (a > b) ? a : b;
}

// ---------------------------------------------------------------------------
// Spatial mean: NHWC (NIMG, 24, 128) f16 -> (NIMG, 128) f32
// ---------------------------------------------------------------------------
__global__ void avg_kernel(const f16* __restrict__ in, float* __restrict__ out)
{
    int idx = blockIdx.x * blockDim.x + threadIdx.x;
    const int total = NIMG * 128;
    if (idx >= total) return;
    int n = idx >> 7;
    int c = idx & 127;
    const f16* p = in + (size_t)n * 24 * 128 + c;
    float s = 0.0f;
#pragma unroll
    for (int i = 0; i < 24; ++i) s += (float)p[(size_t)i * 128];
    out[idx] = s * (1.0f / 24.0f);
}

// ---------------------------------------------------------------------------
// Final linear: out[n][oc] = lb[oc] + sum_c avg[n][c] * lw[oc][c]
// ---------------------------------------------------------------------------
__global__ void final_kernel(const float* __restrict__ avg,
                             const float* __restrict__ lw,
                             const float* __restrict__ lb,
                             float* __restrict__ out)
{
    int idx = blockIdx.x * blockDim.x + threadIdx.x;
    const int total = NIMG * 25;
    if (idx >= total) return;
    int n  = idx / 25;
    int oc = idx - n * 25;
    const float* a = avg + (size_t)n * 128;
    const float* w = lw + (size_t)oc * 128;
    float s = lb[oc];
#pragma unroll 8
    for (int c = 0; c < 128; ++c) s += a[c] * w[c];
    out[idx] = s;
}

// ---------------------------------------------------------------------------
// Host-side orchestration
// ---------------------------------------------------------------------------
struct LayerCfg {
    int Cout, Cin, KH, KW, Kpad;
};

extern "C" void kernel_launch(void* const* d_in, const int* in_sizes, int n_in,
                              void* d_out, int out_size, void* d_ws, size_t ws_size,
                              hipStream_t stream)
{
    (void)in_sizes; (void)n_in; (void)out_size; (void)ws_size;

    const float* x = (const float*)d_in[0];
    // Conv layer leaves (sorted dict keys): base = 1 + 6*l -> [b, beta, gamma, mean, var, w]
    const float* lin_b = (const float*)d_in[43];
    const float* lin_w = (const float*)d_in[44];

    static const LayerCfg cfg[7] = {
        { 32,  1,  3, 3,   32},  // b1[0] (Kreal=9 -> pad 32)
        { 32, 32,  3, 3,  288},  // b1[1]
        { 32, 32,  3, 3,  288},  // b1[2]
        { 32, 32,  3, 3,  288},  // b1[3]
        { 64, 32,  3, 3,  288},  // b2[0]
        { 64, 64,  3, 3,  576},  // b2[1]
        {128, 64, 12, 9, 6912},  // b3[0]
    };

    // ---- workspace carve-up ----
    char* ws = (char*)d_ws;
    size_t off = 0;
    auto carve = [&](size_t bytes) -> char* {
        char* p = ws + off;
        off += (bytes + 255) & ~(size_t)255;
        return p;
    };

    const size_t padBytes = (size_t)NIMG * 86 * 17 * 32 * sizeof(f16);  // halo-padded big buf
    const size_t winBytes = (size_t)NIMG * 86 * 17 * 1 * sizeof(f16);
    f16* pA   = (f16*)carve(padBytes);
    f16* pB   = (f16*)carve(padBytes);
    f16* wbu  = (f16*)carve(winBytes);

    f16*   wbuf[7];
    float* lscale[7];
    float* lbias[7];
    for (int l = 0; l < 7; ++l) {
        wbuf[l]   = (f16*)carve((size_t)cfg[l].Cout * cfg[l].Kpad * sizeof(f16));
        lscale[l] = (float*)carve((size_t)cfg[l].Cout * sizeof(float));
        lbias[l]  = (float*)carve((size_t)cfg[l].Cout * sizeof(float));
    }
    float* avgbuf = (float*)carve((size_t)NIMG * 128 * sizeof(float));

    // ---- zero halo-padded buffers (graph-capture-safe memset nodes) ----
    (void)hipMemsetAsync(pA, 0, padBytes, stream);
    (void)hipMemsetAsync(pB, 0, padBytes, stream);
    (void)hipMemsetAsync(wbu, 0, winBytes, stream);

    // ---- prep all conv layers ----
    for (int l = 0; l < 7; ++l) {
        int base = 1 + 6 * l;
        prep_layer_kernel<<<(cfg[l].Cout * cfg[l].Kpad + 255) / 256, 256, 0, stream>>>(
            (const float*)d_in[base + 5], (const float*)d_in[base + 0],
            (const float*)d_in[base + 1], (const float*)d_in[base + 2],
            (const float*)d_in[base + 3], (const float*)d_in[base + 4],
            wbuf[l], lscale[l], lbias[l],
            cfg[l].Cout, cfg[l].Cin, cfg[l].KH, cfg[l].KW, cfg[l].Kpad);
    }

    // ---- windowed input into padded (86,17,1) buffer ----
    window_kernel<<<(NIMG * 84 * 15 + 255) / 256, 256, 0, stream>>>(x, wbu);

#define CONV(L, IN, OUT, CIN, IH, IW, COUT, OH, OW, OHB, OWB, OOFS, KH, KW, KPAD)  \
    do {                                                                           \
        dim3 grid((NIMG * OH * OW / 32) / 8, COUT / 32);                           \
        conv_wmma_kernel<CIN, IH, IW, COUT, OH, OW, OHB, OWB, OOFS, KH, KW, KPAD>  \
            <<<grid, 256, 0, stream>>>(IN, OUT, wbuf[L], lscale[L], lbias[L]);     \
    } while (0)

    // ---- network ----
    // b1: halo-padded 86x17 buffers (logical 84x15, pad-1 convs)
    CONV(0, wbu, pA,  1, 86, 17, 32, 84, 15, 86, 17, 1, 3, 3,   32);
    CONV(1, pA,  pB, 32, 86, 17, 32, 84, 15, 86, 17, 1, 3, 3,  288);
    CONV(2, pB,  pA, 32, 86, 17, 32, 84, 15, 86, 17, 1, 3, 3,  288);
    CONV(3, pA,  pB, 32, 86, 17, 32, 84, 15, 86, 17, 1, 3, 3,  288);
    // pool1: padded (86,17) interior -> unpadded (42,15,32)
    pool_kernel<32, 84, 15, 86, 17, 1>
        <<<(NIMG * 42 * 15 * 32 + 255) / 256, 256, 0, stream>>>(pB, pA);
    // b2: valid convs, unpadded buffers
    CONV(4, pA, pB, 32, 42, 15, 64, 40, 13, 40, 13, 0, 3, 3,  288);
    CONV(5, pB, pA, 64, 40, 13, 64, 38, 11, 38, 11, 0, 3, 3,  576);
    // pool2: unpadded (38,11,64) -> (19,11,64)
    pool_kernel<64, 38, 11, 38, 11, 0>
        <<<(NIMG * 19 * 11 * 64 + 255) / 256, 256, 0, stream>>>(pA, pB);
    // b3: valid 12x9 conv -> (8,3,128)
    CONV(6, pB, pA, 64, 19, 11, 128, 8, 3, 8, 3, 0, 12, 9, 6912);

    avg_kernel<<<(NIMG * 128 + 255) / 256, 256, 0, stream>>>(pA, avgbuf);
    final_kernel<<<(NIMG * 25 + 255) / 256, 256, 0, stream>>>(
        avgbuf, lin_w, lin_b, (float*)d_out);

#undef CONV
}
